// CSNNet_41601053229874
// MI455X (gfx1250) — matrix operand order repaired and checked by reading, not compiled, not run on hardware
//
#include <hip/hip_runtime.h>

// Fully fused spiking-CNN forward for MI455X (gfx1250, wave32).
// One workgroup per batch element; the whole 32-step recurrent loop lives in
// the kernel; all state (x, mem1, mem2, spk1) resident in LDS (~288 KB of the
// 320 KB/WGP). conv2 channel mixing runs on v_wmma_f32_16x16x32_f16 with f32
// accumulation (spikes are exactly representable in f16). Spikes are staged
// position-major-interleaved so each lane's WMMA B operand is 3 ds_load_b64.

typedef __attribute__((ext_vector_type(16))) _Float16 v16h;
typedef __attribute__((ext_vector_type(4)))  _Float16 v4h;
typedef __attribute__((ext_vector_type(8)))  float    v8f;

#define BETA    0.9f
#define NSTEPS  32
#define LFULL   8192
#define LH      4096   // after maxpool1
#define LQ      2048   // after maxpool2
#define NCH     8
#define THREADS 1024
#define NWAVES  32

__global__ __launch_bounds__(THREADS)
void snn_fused_kernel(const float* __restrict__ x,
                      const float* __restrict__ w1,  const float* __restrict__ b1,
                      const float* __restrict__ w2,  const float* __restrict__ b2,
                      const float* __restrict__ fcw, const float* __restrict__ fcb,
                      const float* __restrict__ th1p, const float* __restrict__ th2p,
                      const float* __restrict__ thop,
                      float* __restrict__ out)
{
    extern __shared__ char smem_raw[];
    float*    x_s   = (float*)smem_raw;             // 8194 (zero halo each side)
    float*    mem1  = x_s  + 8194;                  // 8*4096
    float*    mem2  = mem1 + NCH * LH;              // 8*2048
    float*    wred  = mem2 + NCH * LQ;              // 32 waves * 2 partials
    _Float16* sp    = (_Float16*)(wred + 2*NWAVES); // [4098 rows][8 ich] halves

    const int tid  = threadIdx.x;
    const int b    = blockIdx.x;
    const int lane = tid & 31;
    const int wave = tid >> 5;

    const float th1 = th1p[0], th2 = th2p[0], tho = thop[0];

    // ---------------- init LDS ----------------
    for (int i = tid; i < LFULL; i += THREADS) x_s[i + 1] = x[(size_t)b * LFULL + i];
    if (tid == 0) { x_s[0] = 0.f; x_s[LFULL + 1] = 0.f; }
    for (int i = tid; i < NCH * LH; i += THREADS) mem1[i] = 0.f;
    for (int i = tid; i < NCH * LQ; i += THREADS) mem2[i] = 0.f;
    if (tid < 2 * NCH) {                  // halo rows 0 and 4097, never rewritten
        const int r = (tid < NCH) ? 0 : 4097;
        sp[r * NCH + (tid & 7)] = (_Float16)0;
    }

    // phase-1 ownership: channel = tid>>7, 128 threads cover 4096 positions
    const int   ch1 = tid >> 7;
    const int   pth = tid & 127;
    const float w1a = w1[ch1 * 3 + 0], w1b = w1[ch1 * 3 + 1], w1c = w1[ch1 * 3 + 2];
    const float bb1 = b1[ch1];

    // ---------------- WMMA A operand: conv2 weights, K = ich*4 + tap -------
    // Contiguous K order used by A and B here: lanes 0-15 hold K={0..15}, lanes 16-31
    // hold K={16..31}, for row M = lane&15. tap==3 and och>=8 are pads.
    const int och   = lane & 15;
    const int aichb = (lane < 16) ? 0 : 4;
    v16h A;
#pragma unroll
    for (int h = 0; h < 16; ++h) {
        const int hk  = h;
        const int tap = hk & 3;                 // compile-time
        const int ich = (hk >> 2) + aichb;
        float wv = 0.f;
        if (och < NCH && tap < 3) wv = w2[(och * NCH + ich) * 3 + tap];
        A[h] = (_Float16)wv;
    }
    float bias2[8];
#pragma unroll
    for (int r = 0; r < 8; ++r) bias2[r] = b2[r];

    float mo0 = 0.f, mo1 = 0.f;              // mem_out lives in thread 0's regs
    const float fcb0 = fcb[0], fcb1 = fcb[1];

    __syncthreads();

    const int base0 = wave * 128;     // wave owns conv positions [base0, base0+128)
    const int ncol  = lane & 15;
    const int bgrp  = (lane < 16) ? 0 : 4;   // B: lanes<16 take ich 0-3, else 4-7
    const bool owner = ((lane & 1) == 0) && (lane < 16);

    for (int t = 0; t < NSTEPS; ++t) {
        // ---- phase 1: recompute cur1 = maxpool(conv1(x)), LIF1, emit f16 spikes
#pragma unroll 4
        for (int j = 0; j < 32; ++j) {
            const int p  = pth + j * 128;    // pooled position 0..4095
            const float xa = x_s[2 * p], xb = x_s[2 * p + 1];
            const float xc = x_s[2 * p + 2], xd = x_s[2 * p + 3];
            const float y0 = fmaf(w1a, xa, fmaf(w1b, xb, fmaf(w1c, xc, bb1)));
            const float y1 = fmaf(w1a, xb, fmaf(w1b, xc, fmaf(w1c, xd, bb1)));
            const float cur1 = fmaxf(y0, y1);
            const int idx = ch1 * LH + p;
            float m = mem1[idx];
            const float reset = (m > th1) ? th1 : 0.f;   // reset from PREVIOUS mem
            m = fmaf(BETA, m, cur1) - reset;
            mem1[idx] = m;
            sp[(p + 1) * NCH + ch1] = (_Float16)((m > th1) ? 1.f : 0.f);
        }
        __syncthreads();

        // ---- phase 2: conv2 via WMMA (16 och x 16 pos, K=32), pool, LIF2, FC
        float pa0 = 0.f, pa1 = 0.f;
#pragma unroll 2
        for (int tt = 0; tt < 8; ++tt) {
            const int base = base0 + tt * 16;
            const int pos  = base + ncol;
            // Three aligned 8-byte LDS loads: taps -1,0,+1 (halo rows give 0)
            const v4h r0 = *(const v4h*)(sp + (pos + 0) * NCH + bgrp);
            const v4h r1 = *(const v4h*)(sp + (pos + 1) * NCH + bgrp);
            const v4h r2 = *(const v4h*)(sp + (pos + 2) * NCH + bgrp);
            v16h B;
#pragma unroll
            for (int ic = 0; ic < 4; ++ic) {      // K = ich*4 + tap, tap 3 = pad
                B[ic * 4 + 0] = r0[ic];
                B[ic * 4 + 1] = r1[ic];
                B[ic * 4 + 2] = r2[ic];
                B[ic * 4 + 3] = (_Float16)0;
            }
            v8f acc = {};
            acc = __builtin_amdgcn_wmma_f32_16x16x32_f16(
                      false, A, false, B, (short)0, acc, false, false);

            // maxpool2 over lane pairs — executed by ALL lanes (uniform)
            float pooled[8];
#pragma unroll
            for (int r = 0; r < 8; ++r)
                pooled[r] = fmaxf(acc[r], __shfl_xor(acc[r], 1, 32));

            // single divergent region per tile: LIF2 + FC partials
            if (owner) {
                const int pp = (base >> 1) + (lane >> 1);
#pragma unroll
                for (int r = 0; r < 8; ++r) {
                    const int midx = r * LQ + pp;     // == flat fc index och*2048+pp
                    const float cur2 = pooled[r] + bias2[r];
                    float m = mem2[midx];
                    const float reset = (m > th2) ? th2 : 0.f;
                    m = fmaf(BETA, m, cur2) - reset;
                    mem2[midx] = m;
                    const float spk2 = (m > th2) ? 1.f : 0.f;
                    pa0 = fmaf(spk2, fcw[midx],         pa0);
                    pa1 = fmaf(spk2, fcw[16384 + midx], pa1);
                }
            }
        }

        // ---- phase 3: deterministic reduction + LIF out + record
#pragma unroll
        for (int off = 16; off > 0; off >>= 1) {
            pa0 += __shfl_xor(pa0, off, 32);
            pa1 += __shfl_xor(pa1, off, 32);
        }
        if (lane == 0) { wred[wave * 2] = pa0; wred[wave * 2 + 1] = pa1; }
        __syncthreads();   // also guarantees all sp reads done before next phase 1
        if (tid == 0) {
            float c0 = fcb0, c1 = fcb1;
            for (int wv = 0; wv < NWAVES; ++wv) { c0 += wred[wv * 2]; c1 += wred[wv * 2 + 1]; }
            const float r0 = (mo0 > tho) ? tho : 0.f;
            const float r1 = (mo1 > tho) ? tho : 0.f;
            mo0 = fmaf(BETA, mo0, c0) - r0;
            mo1 = fmaf(BETA, mo1, c1) - r1;
            out[t * 512 + b * 2 + 0]         = (mo0 > tho) ? 1.f : 0.f;  // spk_rec
            out[t * 512 + b * 2 + 1]         = (mo1 > tho) ? 1.f : 0.f;
            out[16384 + t * 512 + b * 2 + 0] = mo0;                      // mem_rec
            out[16384 + t * 512 + b * 2 + 1] = mo1;
        }
    }
}

extern "C" void kernel_launch(void* const* d_in, const int* in_sizes, int n_in,
                              void* d_out, int out_size, void* d_ws, size_t ws_size,
                              hipStream_t stream) {
    (void)n_in; (void)d_ws; (void)ws_size; (void)out_size;
    const float* x   = (const float*)d_in[0];
    const float* w1  = (const float*)d_in[1];
    const float* b1  = (const float*)d_in[2];
    const float* w2  = (const float*)d_in[3];
    const float* b2  = (const float*)d_in[4];
    const float* fcw = (const float*)d_in[5];
    const float* fcb = (const float*)d_in[6];
    const float* t1  = (const float*)d_in[7];
    const float* t2  = (const float*)d_in[8];
    const float* to  = (const float*)d_in[9];
    float* out = (float*)d_out;

    const int B = in_sizes[0] / LFULL;   // 256
    const size_t shmem = (size_t)(8194 + NCH * LH + NCH * LQ + 2 * NWAVES) * sizeof(float)
                       + (size_t)(LH + 2) * NCH * sizeof(_Float16);   // ~295 KB of 320 KB

    hipFuncSetAttribute((const void*)snn_fused_kernel,
                        hipFuncAttributeMaxDynamicSharedMemorySize, (int)shmem);
    snn_fused_kernel<<<dim3(B), dim3(THREADS), shmem, stream>>>(
        x, w1, b1, w2, b2, fcw, fcb, t1, t2, to, out);
}
